// LowRankSelfAttention_86921548136920
// MI455X (gfx1250) — matrix operand order repaired and checked by reading, not compiled, not run on hardware
//
#include <hip/hip_runtime.h>

// ---------------------------------------------------------------------------
// CDNA5 (gfx1250) wave32 WMMA low-rank self-attention.
//   B=4, S=4096, D=1024, R=32.  All GEMMs: v_wmma_f32_16x16x32_bf16.
// All WMMA operands staged in memory as bf16 in fragment-natural order so
// every fragment load is 2x global_load_b128 / ds_load_b128.
// Attention K/V tiles are staged with async global->LDS DMA (ASYNCcnt).
// ---------------------------------------------------------------------------

typedef __attribute__((ext_vector_type(16))) __bf16 v16bf;
typedef __attribute__((ext_vector_type(8)))  float  v8f;
typedef __attribute__((ext_vector_type(4)))  float  f4;

#define EMBED_DIM 1024
#define SEQ 4096
#define NBATCH 4
#define RANK 32
#define ROWS (NBATCH * SEQ)   // 16384 flat rows

__device__ __forceinline__ __bf16 f2bf(float x) { return (__bf16)x; }

union V16 { v16bf v; f4 q[2]; };

// Load a 16-element bf16 fragment from two 16-byte-aligned 8-element runs.
__device__ __forceinline__ v16bf load2x16B(const __bf16* p0, const __bf16* p1) {
  V16 u;
  u.q[0] = *(const f4*)p0;
  u.q[1] = *(const f4*)p1;
  return u.v;
}

// A-frag from an f32 row (x): 4x b128 loads + cvt to bf16.
__device__ __forceinline__ v16bf a_frag_f32(const float* __restrict__ row, int hf) {
  const f4* p  = (const f4*)(row + hf * 16);
  const f4* p2 = (const f4*)(row + hf * 16 + 8);
  f4 a0 = p[0], a1 = p[1], b0 = p2[0], b1 = p2[1];
  v16bf r;
#pragma unroll
  for (int i = 0; i < 4; ++i) {
    r[i]      = f2bf(a0[i]);
    r[4 + i]  = f2bf(a1[i]);
    r[8 + i]  = f2bf(b0[i]);
    r[12 + i] = f2bf(b1[i]);
  }
  return r;
}

__device__ __forceinline__ v8f wmma_bf16(v16bf a, v16bf b, v8f c) {
  return __builtin_amdgcn_wmma_f32_16x16x32_bf16(
      false, a, false, b, (short)0, c, false, false);
}

// Async global -> LDS 16-byte copy (gfx1250 GLOBAL_LOAD_ASYNC_TO_LDS_B128,
// tracked by ASYNCcnt).  lds_addr = LDS byte address (generic addr low 32b).
__device__ __forceinline__ void async_ld_b128(unsigned lds_addr, const void* gaddr) {
  asm volatile("global_load_async_to_lds_b128 %0, %1, off"
               :: "v"(lds_addr), "v"(gaddr) : "memory");
}
__device__ __forceinline__ void wait_async0() {
  asm volatile("s_wait_asynccnt 0x0" ::: "memory");
}

// ---------------------------------------------------------------------------
// Kernel 0: weight prep.  Wq/Wk/Wv [D x R]f32 -> [R x D]bf16 (transposed),
//           Wo [R x D]f32 -> [D x R]bf16 (transposed).
// ---------------------------------------------------------------------------
__global__ __launch_bounds__(256) void prep_kernel(
    const float* __restrict__ Wq, const float* __restrict__ Wk,
    const float* __restrict__ Wv, const float* __restrict__ Wo,
    __bf16* __restrict__ WqT, __bf16* __restrict__ WkT,
    __bf16* __restrict__ WvT, __bf16* __restrict__ WoT)
{
  int i = blockIdx.x * 256 + threadIdx.x;    // 0 .. 32767
  int r = i & 31, d = i >> 5;
  WqT[(size_t)r * EMBED_DIM + d] = f2bf(Wq[(size_t)d * RANK + r]);
  WkT[(size_t)r * EMBED_DIM + d] = f2bf(Wk[(size_t)d * RANK + r]);
  WvT[(size_t)r * EMBED_DIM + d] = f2bf(Wv[(size_t)d * RANK + r]);
  WoT[(size_t)d * RANK + r]      = f2bf(Wo[(size_t)r * EMBED_DIM + d]);
}

// ---------------------------------------------------------------------------
// Kernel 1: Q/K/V projection.  One pass over x.
//   Qb [16384x32] bf16 (pre-scaled by R^-1/2), Kb [16384x32] bf16,
//   Vt [B][32][4096] bf16 (transposed for the P@V B-fragment).
// ---------------------------------------------------------------------------
__global__ __launch_bounds__(256) void qkv_kernel(
    const float* __restrict__ x,
    const __bf16* __restrict__ WqT, const float* __restrict__ bq,
    const __bf16* __restrict__ WkT, const float* __restrict__ bk,
    const __bf16* __restrict__ WvT, const float* __restrict__ bv,
    __bf16* __restrict__ Qb, __bf16* __restrict__ Kb, __bf16* __restrict__ Vt)
{
  const int lane = threadIdx.x & 31;
  const int wave = threadIdx.x >> 5;
  const int tile = blockIdx.x * 8 + wave;      // 1024 tiles of 16 rows
  const int row0 = tile * 16;
  const int hf   = lane >> 4;
  const int mn   = lane & 15;

  const float*  xrow = x + (size_t)(row0 + mn) * EMBED_DIM;
  const __bf16* wq0 = WqT + (size_t)mn * EMBED_DIM;
  const __bf16* wq1 = WqT + (size_t)(mn + 16) * EMBED_DIM;
  const __bf16* wk0 = WkT + (size_t)mn * EMBED_DIM;
  const __bf16* wk1 = WkT + (size_t)(mn + 16) * EMBED_DIM;
  const __bf16* wv0 = WvT + (size_t)mn * EMBED_DIM;
  const __bf16* wv1 = WvT + (size_t)(mn + 16) * EMBED_DIM;

  v8f accQ0 = {}, accQ1 = {}, accK0 = {}, accK1 = {}, accV0 = {}, accV1 = {};

  for (int kc = 0; kc < EMBED_DIM; kc += 32) {
    __builtin_prefetch(xrow + kc + 128, 0, 1);        // global_prefetch_b8
    v16bf a = a_frag_f32(xrow + kc, hf);
    const int ko = kc + hf * 16;                      // B-frag K-run start
    v16bf fq0 = load2x16B(wq0 + ko, wq0 + ko + 8);
    v16bf fq1 = load2x16B(wq1 + ko, wq1 + ko + 8);
    v16bf fk0 = load2x16B(wk0 + ko, wk0 + ko + 8);
    v16bf fk1 = load2x16B(wk1 + ko, wk1 + ko + 8);
    v16bf fv0 = load2x16B(wv0 + ko, wv0 + ko + 8);
    v16bf fv1 = load2x16B(wv1 + ko, wv1 + ko + 8);
    accQ0 = wmma_bf16(a, fq0, accQ0);  accQ1 = wmma_bf16(a, fq1, accQ1);
    accK0 = wmma_bf16(a, fk0, accK0);  accK1 = wmma_bf16(a, fk1, accK1);
    accV0 = wmma_bf16(a, fv0, accV0);  accV1 = wmma_bf16(a, fv1, accV1);
  }

  const float scale = 0.17677669529663687f;           // RANK^-0.5
  const int   b = row0 >> 12;                         // batch of this tile
#pragma unroll
  for (int r = 0; r < 8; ++r) {
    int row = row0 + r + hf * 8;
    int s   = row & (SEQ - 1);
    size_t o = (size_t)row * RANK + mn;
    Qb[o]      = f2bf((accQ0[r] + bq[mn]) * scale);
    Qb[o + 16] = f2bf((accQ1[r] + bq[mn + 16]) * scale);
    Kb[o]      = f2bf(accK0[r] + bk[mn]);
    Kb[o + 16] = f2bf(accK1[r] + bk[mn + 16]);
    Vt[(size_t)b * RANK * SEQ + (size_t)mn * SEQ + s]        = f2bf(accV0[r] + bv[mn]);
    Vt[(size_t)b * RANK * SEQ + (size_t)(mn + 16) * SEQ + s] = f2bf(accV1[r] + bv[mn + 16]);
  }
}

// ---------------------------------------------------------------------------
// Kernel 2: flash attention.  128 threads = 4 waves per block; each wave owns
// 32 query rows (two 16-row A-frags).  K/V 32-key tiles are staged into
// double-buffered LDS with async global->LDS DMA, shared by all 4 waves; one
// s_wait_asynccnt + __syncthreads per key block (uniform trip count).
// ---------------------------------------------------------------------------
__global__ __launch_bounds__(128) void attn_kernel(
    const __bf16* __restrict__ Qb, const __bf16* __restrict__ Kb,
    const __bf16* __restrict__ Vt, __bf16* __restrict__ AOb)
{
  __shared__ __align__(16) __bf16 lk[2][32][32];     // [buf][key][r]
  __shared__ __align__(16) __bf16 lv[2][32][32];     // [buf][r][key]
  __shared__ __align__(16) __bf16 lp[4][2][16][40];  // [wave][qfrag][row][key]

  const int tid  = threadIdx.x;
  const int lane = tid & 31;
  const int wave = tid >> 5;
  const int blk  = blockIdx.x;                // 128 blocks; 32 per batch
  const int b    = blk >> 5;
  const int q0   = ((blk & 31) * 4 + wave) * 32;  // wave's first query in batch
  const int hf   = lane >> 4;
  const int mn   = lane & 15;
  const size_t rbase = (size_t)b * SEQ;
  const size_t vbase = (size_t)b * RANK * SEQ;

  const int vrow  = tid >> 2;                 // V staging row (0..31)
  const int vpart = tid & 3;                  // 16B chunk within row
  const unsigned lk_lds0 = (unsigned)(uintptr_t)&lk[0][0][0] + tid * 16;
  const unsigned lk_lds1 = (unsigned)(uintptr_t)&lk[1][0][0] + tid * 16;
  const unsigned lv_lds0 = (unsigned)(uintptr_t)&lv[0][vrow][0] + vpart * 16;
  const unsigned lv_lds1 = (unsigned)(uintptr_t)&lv[1][vrow][0] + vpart * 16;

  // stage one 32-key K tile (2KB) + V tile (2KB): two async b128 DMAs/thread
  auto stage = [&](int buf, int j) {
    async_ld_b128(buf ? lk_lds1 : lk_lds0,
                  (const char*)(Kb + (rbase + j) * RANK) + tid * 16);
    async_ld_b128(buf ? lv_lds1 : lv_lds0,
                  (const char*)(Vt + vbase + (size_t)vrow * SEQ + j) + vpart * 16);
  };

  // Q A-frags (already scaled by R^-1/2 in qkv_kernel)
  v16bf qa[2];
#pragma unroll
  for (int g = 0; g < 2; ++g) {
    const __bf16* qrow = Qb + (rbase + q0 + g * 16 + mn) * RANK;
    qa[g] = load2x16B(qrow + hf * 16, qrow + hf * 16 + 8);
  }

  v8f oc[2][2] = {{{}, {}}, {{}, {}}};
  float mi[2][8], li[2][8];
#pragma unroll
  for (int g = 0; g < 2; ++g)
#pragma unroll
    for (int r = 0; r < 8; ++r) { mi[g][r] = -1e30f; li[g][r] = 0.0f; }

  stage(0, 0);
  wait_async0();
  __syncthreads();

  for (int j = 0; j < SEQ; j += 32) {
    const int buf = (j >> 5) & 1;
    if (j + 32 < SEQ) stage(buf ^ 1, j + 32);   // async-prefetch next tile

    // K^T B-frags from LDS: column = key (ntile*16+mn), K-run over r
    const __bf16* kc0 = &lk[buf][mn][hf * 16];
    const __bf16* kc1 = &lk[buf][mn + 16][hf * 16];
    v16bf kb0 = load2x16B(kc0, kc0 + 8);
    v16bf kb1 = load2x16B(kc1, kc1 + 8);
    // V B-frags from LDS: column = r (mn / mn+16), K-run over keys
    const __bf16* vc0 = &lv[buf][mn][hf * 16];
    const __bf16* vc1 = &lv[buf][mn + 16][hf * 16];
    v16bf vb0 = load2x16B(vc0, vc0 + 8);
    v16bf vb1 = load2x16B(vc1, vc1 + 8);

    v8f z = {};
    v8f s[2][2];
    s[0][0] = wmma_bf16(qa[0], kb0, z);
    s[0][1] = wmma_bf16(qa[0], kb1, z);
    s[1][0] = wmma_bf16(qa[1], kb0, z);
    s[1][1] = wmma_bf16(qa[1], kb1, z);

#pragma unroll
    for (int g = 0; g < 2; ++g) {
#pragma unroll
      for (int r = 0; r < 8; ++r) {
        float mx = fmaxf(s[g][0][r], s[g][1][r]);
        mx = fmaxf(mx, __shfl_xor(mx, 1));
        mx = fmaxf(mx, __shfl_xor(mx, 2));
        mx = fmaxf(mx, __shfl_xor(mx, 4));
        mx = fmaxf(mx, __shfl_xor(mx, 8));
        float mnew  = fmaxf(mi[g][r], mx);
        float alpha = __expf(mi[g][r] - mnew);
        float p0 = __expf(s[g][0][r] - mnew);
        float p1 = __expf(s[g][1][r] - mnew);
        float rs = p0 + p1;
        rs += __shfl_xor(rs, 1);
        rs += __shfl_xor(rs, 2);
        rs += __shfl_xor(rs, 4);
        rs += __shfl_xor(rs, 8);
        li[g][r] = li[g][r] * alpha + rs;
        mi[g][r] = mnew;
        oc[g][0][r] *= alpha;
        oc[g][1][r] *= alpha;
        lp[wave][g][r + hf * 8][mn]      = f2bf(p0);
        lp[wave][g][r + hf * 8][mn + 16] = f2bf(p1);
      }
    }

    // P (C-layout -> A-layout via per-wave LDS region), then P@V
#pragma unroll
    for (int g = 0; g < 2; ++g) {
      const __bf16* lrow = &lp[wave][g][mn][0];
      v16bf pa = load2x16B(lrow + hf * 16, lrow + hf * 16 + 8);
      oc[g][0] = wmma_bf16(pa, vb0, oc[g][0]);
      oc[g][1] = wmma_bf16(pa, vb1, oc[g][1]);
    }

    wait_async0();     // this thread's async DMAs into buf^1 are complete
    __syncthreads();   // tile consumed by all waves; next tile visible
  }

#pragma unroll
  for (int g = 0; g < 2; ++g)
#pragma unroll
    for (int r = 0; r < 8; ++r) {
      float inv = 1.0f / li[g][r];
      size_t o = (rbase + q0 + g * 16 + r + hf * 8) * RANK + mn;
      AOb[o]      = f2bf(oc[g][0][r] * inv);
      AOb[o + 16] = f2bf(oc[g][1][r] * inv);
    }
}

// ---------------------------------------------------------------------------
// Kernel 3: output projection.  AO[16384x32]bf16 * WoT + bo -> out f32.
// ---------------------------------------------------------------------------
__global__ __launch_bounds__(256) void outproj_kernel(
    const __bf16* __restrict__ AOb, const __bf16* __restrict__ WoT,
    const float* __restrict__ bo, float* __restrict__ out)
{
  const int lane = threadIdx.x & 31;
  const int wave = threadIdx.x >> 5;
  const int tile = blockIdx.x * 8 + wave;   // 1024 row tiles
  const int row0 = tile * 16;
  const int hf   = lane >> 4;
  const int mn   = lane & 15;

  const __bf16* arow = AOb + (size_t)(row0 + mn) * RANK;
  v16bf a = load2x16B(arow + hf * 16, arow + hf * 16 + 8);

  for (int nt = 0; nt < EMBED_DIM / 16; ++nt) {
    const __bf16* w = WoT + (size_t)(nt * 16 + mn) * RANK + hf * 16;
    v16bf bf = load2x16B(w, w + 8);
    v8f z = {};
    v8f acc = wmma_bf16(a, bf, z);
    float bias = bo[nt * 16 + mn];
#pragma unroll
    for (int r = 0; r < 8; ++r) {
      out[(size_t)(row0 + r + hf * 8) * EMBED_DIM + nt * 16 + mn] = acc[r] + bias;
    }
  }
}

// ---------------------------------------------------------------------------
extern "C" void kernel_launch(void* const* d_in, const int* in_sizes, int n_in,
                              void* d_out, int out_size, void* d_ws, size_t ws_size,
                              hipStream_t stream) {
  const float* x  = (const float*)d_in[0];
  const float* Wq = (const float*)d_in[1];
  const float* bq = (const float*)d_in[2];
  const float* Wk = (const float*)d_in[3];
  const float* bk = (const float*)d_in[4];
  const float* Wv = (const float*)d_in[5];
  const float* bv = (const float*)d_in[6];
  const float* Wo = (const float*)d_in[7];
  const float* bo = (const float*)d_in[8];
  float* out = (float*)d_out;

  // bf16 workspace (~4.25 MB)
  __bf16* ws  = (__bf16*)d_ws;
  __bf16* Qb  = ws;                                  // 16384*32
  __bf16* Kb  = Qb  + (size_t)ROWS * RANK;
  __bf16* Vt  = Kb  + (size_t)ROWS * RANK;           // [B][32][4096]
  __bf16* AOb = Vt  + (size_t)ROWS * RANK;
  __bf16* WqT = AOb + (size_t)ROWS * RANK;           // [32][1024]
  __bf16* WkT = WqT + (size_t)RANK * EMBED_DIM;
  __bf16* WvT = WkT + (size_t)RANK * EMBED_DIM;
  __bf16* WoT = WvT + (size_t)RANK * EMBED_DIM;      // [1024][32]

  prep_kernel<<<128, 256, 0, stream>>>(Wq, Wk, Wv, Wo, WqT, WkT, WvT, WoT);
  qkv_kernel<<<128, 256, 0, stream>>>(x, WqT, bq, WkT, bk, WvT, bv, Qb, Kb, Vt);
  attn_kernel<<<128, 128, 0, stream>>>(Qb, Kb, Vt, AOb);
  outproj_kernel<<<128, 256, 0, stream>>>(AOb, WoT, bo, out);
}
